// LovaszSoftmaxLoss_29884382445875
// MI455X (gfx1250) — compile-verified, matrix operand order checked by reading, not run on hardware
//
#include <hip/hip_runtime.h>
#include <hip/hip_bf16.h>

// Lovasz-Softmax for MI455X (gfx1250, wave32).
// - Pass 1: fused softmax + per-class error histograms (counting-sort replaces
//   19 x 2M-element sorts). HBM-bound: one 160MB read + L2 atomics.
// - Pass 2 (per class): TDM bulk copy of the 3 histogram arrays global->LDS
//   (tensor_load_to_lds, TENSORcnt), prefix scans (f32-exact), Jaccard weights,
//   and the weighted reduction on the matrix pipe via v_wmma_f32_16x16x32_f16
//   (diagonal-dot trick).

#define NCLS 19
#define BB   8
#define HH   512
#define WW   512
#define HWP  (HH * WW)
#define NPIX (BB * HWP)
#define NB   4096            // error buckets per class

typedef __attribute__((ext_vector_type(16))) _Float16 v16h;
typedef __attribute__((ext_vector_type(8)))  float    v8f;
typedef __attribute__((ext_vector_type(4)))  unsigned int v4u;
typedef __attribute__((ext_vector_type(8)))  int      v8i;
typedef __attribute__((ext_vector_type(4)))  int      v4i;

#if defined(__AMDGCN__) && __has_builtin(__builtin_amdgcn_tensor_load_to_lds)
#define USE_TDM 1
#else
#define USE_TDM 0
#endif

// ---------------------------------------------------------------- zero scratch
__global__ void lovasz_zero_ws(unsigned int* __restrict__ w, int nwords) {
  int i = blockIdx.x * blockDim.x + threadIdx.x;
  if (i < nwords) w[i] = 0u;
}

// ------------------------------------------------- pass 1: softmax + histogram
__global__ __launch_bounds__(256)
void lovasz_hist(const float* __restrict__ logits,
                 const long long* __restrict__ target,
                 unsigned int* __restrict__ histN,
                 unsigned int* __restrict__ histF,
                 float* __restrict__ histS) {
  int p = blockIdx.x * blockDim.x + threadIdx.x;
  if (p >= NPIX) return;
  int b  = p / HWP;
  int hw = p - b * HWP;
  const float* base = logits + (size_t)b * NCLS * HWP + hw;
  int tgt = (int)target[p];

  // Strided class columns: prefetch ahead (global_prefetch_b8 on gfx1250).
  __builtin_prefetch(base + (size_t)4  * HWP, 0, 1);
  __builtin_prefetch(base + (size_t)10 * HWP, 0, 1);
  __builtin_prefetch(base + (size_t)16 * HWP, 0, 1);

  float l[NCLS];
  float m = -3.0e38f;
#pragma unroll
  for (int c = 0; c < NCLS; ++c) {
    float v = base[(size_t)c * HWP];
    l[c] = v;
    m = fmaxf(m, v);
  }
  float s = 0.f;
#pragma unroll
  for (int c = 0; c < NCLS; ++c) { l[c] = __expf(l[c] - m); s += l[c]; }
  float inv = 1.0f / s;

#pragma unroll
  for (int c = 0; c < NCLS; ++c) {
    float prob = l[c] * inv;
    bool  fg   = (c == tgt);
    float err  = fg ? (1.0f - prob) : prob;
    int bkt = (int)(err * (float)NB);
    bkt = bkt < 0 ? 0 : (bkt > NB - 1 ? NB - 1 : bkt);
    int idx = c * NB + bkt;
    atomicAdd(&histN[idx], 1u);
    if (fg) atomicAdd(&histF[idx], 1u);
    atomicAdd(&histS[idx], err);
  }
}

// --------------------------------------------------------- TDM 1-D LDS copy
// Build a D# (ISA ch.8) describing a contiguous 1-D copy of `ndwords` dwords
// from `gaddr` into LDS byte address `lds_byte`, and issue TENSOR_LOAD_TO_LDS.
#if USE_TDM
__device__ __forceinline__ void tdm_load_1d(unsigned lds_byte, const void* gaddr,
                                            unsigned ndwords) {
  unsigned long long ga = (unsigned long long)(size_t)gaddr;
  v4u g0;
  g0[0] = 1u;                                              // count=1, user D#
  g0[1] = lds_byte;                                        // lds_addr
  g0[2] = (unsigned)(ga & 0xFFFFFFFFu);                    // global_addr[31:0]
  g0[3] = (unsigned)((ga >> 32) & 0x01FFFFFFu) | (2u << 30); // addr[56:32]|type=2
  v8i g1;
  g1[0] = (int)(2u << 16);               // workgroup_mask=0, data_size=2 (4B)
  g1[1] = (int)((ndwords & 0xFFFFu) << 16);   // tensor_dim0[15:0] @ bits 63:48
  g1[2] = (int)(((ndwords >> 16) & 0xFFFFu) | (1u << 16)); // dim0 hi | tensor_dim1=1
  g1[3] = (int)(ndwords << 16);          // tile_dim0 @ bits 127:112
  g1[4] = 1;                             // tile_dim1=1, tile_dim2=0
  g1[5] = (int)ndwords;                  // tensor_dim0_stride[31:0]
  g1[6] = 0;                             // stride hi, tensor_dim1_stride lo
  g1[7] = 0;
  v4i g2 = {1, 1, 0, 0};                 // tensor_dim2=1, tensor_dim3=1
  v4i g3 = {0, 0, 0, 0};
#if __clang_major__ >= 23
  v8i gz = {0, 0, 0, 0, 0, 0, 0, 0};
  __builtin_amdgcn_tensor_load_to_lds(g0, g1, g2, g3, gz, 0);
#else
  __builtin_amdgcn_tensor_load_to_lds(g0, g1, g2, g3, 0);
#endif
}
#endif

// ---------------------------- pass 2: per-class scan + jaccard + WMMA dot
__global__ __launch_bounds__(256)
void lovasz_perclass(const unsigned int* __restrict__ histN,
                     const unsigned int* __restrict__ histF,
                     const float* __restrict__ histS,
                     float* __restrict__ accum) {
  __shared__ float sN[NB];          // raw u32 bits -> cumulative count (f32)
  __shared__ float sF[NB];          // raw u32 bits -> cumulative fg -> jaccard
  __shared__ float sS[NB];          // raw err sums -> reversed err sums
  __shared__ float tN[256], tF[256], tS[256];
  __shared__ int   topRb[256];
  __shared__ float topNN[256], topFF[256], topSS[256];
  __shared__ float shG, shS, shJ0, shE1, shDot;

  const int c   = blockIdx.x;
  const int tid = threadIdx.x;
  const int PER = NB / 256;         // 16 buckets per thread
  const int base = tid * PER;

  // ---- Stage raw per-class histograms into LDS (original bucket order).
#if USE_TDM
  if (tid < 32) {                   // one TDM issue per block (wave 0)
    tdm_load_1d((unsigned)(size_t)(const void*)&sN[0], histN + (size_t)c * NB, NB);
    tdm_load_1d((unsigned)(size_t)(const void*)&sF[0], histF + (size_t)c * NB, NB);
    tdm_load_1d((unsigned)(size_t)(const void*)&sS[0], histS + (size_t)c * NB, NB);
    __builtin_amdgcn_s_wait_tensorcnt(0);
  }
#else
  for (int i = 0; i < PER; ++i) {
    int o = base + i;
    sN[o] = __uint_as_float(histN[c * NB + o]);
    sF[o] = __uint_as_float(histF[c * NB + o]);
    sS[o] = histS[c * NB + o];
  }
#endif
  __syncthreads();

  // ---- Read reversed raw block into registers (rb=0 <-> largest error).
  float rn[PER], rf[PER], rs[PER];
  int   lmin = NB;
  float ln = 0.f, lf = 0.f, ls = 0.f;
  for (int i = 0; i < PER; ++i) {
    int rb   = base + i;
    int orig = NB - 1 - rb;
    unsigned un = __float_as_uint(sN[orig]);
    unsigned uf = __float_as_uint(sF[orig]);
    float    s  = sS[orig];
    rn[i] = (float)un; rf[i] = (float)uf; rs[i] = s;
    if (un > 0u && lmin == NB) { lmin = rb; ln = rn[i]; lf = rf[i]; ls = s; }
  }
  topRb[tid] = lmin; topNN[tid] = ln; topFF[tid] = lf; topSS[tid] = ls;
  __syncthreads();

  // ---- Write converted+reversed values; per-thread inclusive chunk scan.
  float aN = 0.f, aF = 0.f, aS = 0.f;
  for (int i = 0; i < PER; ++i) {
    aN += rn[i]; sN[base + i] = aN;
    aF += rf[i]; sF[base + i] = aF;
    aS += rs[i]; sS[base + i] = rs[i];
  }
  tN[tid] = aN; tF[tid] = aF; tS[tid] = aS;
  __syncthreads();

  // ---- Thread 0: exclusive scan of chunk totals + global stats + j0/e1.
  if (tid == 0) {
    float cn = 0.f, cf = 0.f, cs = 0.f;
    int   tb = NB; float bn = 0.f, bf = 0.f, bs = 0.f;
    for (int t = 0; t < 256; ++t) {
      float nn = tN[t], ff = tF[t];
      tN[t] = cn; tF[t] = cf;
      cn += nn; cf += ff; cs += tS[t];
      if (topRb[t] < tb) { tb = topRb[t]; bn = topNN[t]; bf = topFF[t]; bs = topSS[t]; }
    }
    float G = cf;                          // gts (total fg for this class)
    shG = G; shS = cs;
    float e1  = (bn > 0.f) ? bs / bn : 0.f;
    float fg1 = (bf > 0.f) ? 1.f : 0.f;
    shJ0 = (G > 0.f) ? (1.f - (G - fg1) / (G + 1.f - fg1)) : 0.f;
    shE1 = e1;
    shDot = 0.f;
  }
  __syncthreads();

  // ---- Inclusive cumulatives + per-bucket Jaccard weight.
  const float offN = tN[tid], offF = tF[tid];
  const float G = shG;
  for (int i = 0; i < PER; ++i) {
    int rb = base + i;
    float I = sN[rb] + offN;               // cumulative count (position)
    float F = sF[rb] + offF;               // cumulative fg
    sN[rb] = I;
    float uni = G + I - F;
    float jac = (uni > 0.f) ? (1.f - (G - F) / uni) : 0.f;
    sF[rb] = jac;
  }
  __syncthreads();

  // ---- Wave 0: loss_c = sum_b sS[b]*sF[b] via v_wmma_f32_16x16x32_f16.
  // A[i][k]=s[cb+32i+k], B[k][j]=jacc[cb+32j+k]; D diagonal = per-chunk dots.
  if (tid < 32) {
    const int L = tid;
    const int M = L & 15;
    const int kbase = (L < 16) ? 0 : 8;    // ISA 7.12.2 16-bit A layout
    v8f acc = {};
    for (int t = 0; t < NB / 512; ++t) {   // 8 WMMAs per class
      int cb = t * 512;
      v16h a, b;
#pragma unroll
      for (int v = 0; v < 8; ++v) {
        int kk = (v < 4) ? (kbase + 2 * v) : (16 + kbase + 2 * (v - 4));
        a[2 * v]     = (_Float16)sS[cb + 32 * M + kk];
        a[2 * v + 1] = (_Float16)sS[cb + 32 * M + kk + 1];
        b[2 * v]     = (_Float16)sF[cb + 32 * M + kk];
        b[2 * v + 1] = (_Float16)sF[cb + 32 * M + kk + 1];
      }
      acc = __builtin_amdgcn_wmma_f32_16x16x32_f16(
          /*neg_a=*/false, a, /*neg_b=*/false, b,
          /*c_mod=*/(short)0, acc, /*reuse_a=*/false, /*reuse_b=*/false);
    }
    // D diagonal (16x16 f32 C/D layout): lanes 0-7 comp L; lanes 24-31 comp L-24.
    float contrib = 0.f;
    if (L < 8)        contrib = acc[L];
    else if (L >= 24) contrib = acc[L - 24];
    atomicAdd(&shDot, contrib);
  }
  __syncthreads();

  if (tid == 0 && shG > 0.f) {
    float lossC = shDot - shJ0 * (shS - shE1);
    atomicAdd(&accum[0], lossC);
    atomicAdd(&accum[1], 1.0f);            // present-class counter
  }
}

// ---------------------------------------------------------------- finalize
__global__ void lovasz_finalize(const float* __restrict__ accum,
                                float* __restrict__ out) {
  float p = accum[1];
  out[0] = (p > 0.f) ? accum[0] / p : 0.f;
}

// ---------------------------------------------------------------- launcher
extern "C" void kernel_launch(void* const* d_in, const int* in_sizes, int n_in,
                              void* d_out, int out_size, void* d_ws, size_t ws_size,
                              hipStream_t stream) {
  const float*     logits = (const float*)d_in[0];
  const long long* target = (const long long*)d_in[1];

  unsigned int* histN = (unsigned int*)d_ws;
  unsigned int* histF = histN + (size_t)NCLS * NB;
  float*        histS = (float*)(histF + (size_t)NCLS * NB);
  float*        accum = histS + (size_t)NCLS * NB;

  int nwords = NCLS * NB * 3 + 2;
  lovasz_zero_ws<<<(nwords + 255) / 256, 256, 0, stream>>>((unsigned int*)d_ws, nwords);
  lovasz_hist<<<(NPIX + 255) / 256, 256, 0, stream>>>(logits, target, histN, histF, histS);
  lovasz_perclass<<<NCLS, 256, 0, stream>>>(histN, histF, histS, accum);
  lovasz_finalize<<<1, 1, 0, stream>>>(accum, (float*)d_out);
}
